// VectorQuantizer_85615878078787
// MI455X (gfx1250) — matrix operand order repaired and checked by reading, not compiled, not run on hardware
//
#include <hip/hip_runtime.h>

// Problem constants (from reference): B=32, C=256, H=32, W=32, K=1024
#define KCODES 1024
#define CDIM   256
#define NVEC   32768          // B*H*W
#define ZQ_ELEMS 8388608      // B*W*C*H

typedef __attribute__((ext_vector_type(16))) __bf16 v16bf;
typedef __attribute__((ext_vector_type(8)))  float  v8f;

static __device__ __forceinline__ unsigned short f32_to_bf16_rn(float f) {
    unsigned int u = __float_as_uint(f);
    u += 0x7fffu + ((u >> 16) & 1u);       // round-to-nearest-even
    return (unsigned short)(u >> 16);
}
static __device__ __forceinline__ float bf16_to_f32(unsigned short h) {
    return __uint_as_float(((unsigned int)h) << 16);
}

// Build a v16bf fragment from two 16-byte chunks at p and p+gap (elements are
// consecutive bf16 -> packed pairs per VGPR, matching the ISA A/B layouts).
static __device__ __forceinline__ v16bf load_frag(const unsigned short* p, int gap) {
    union { v16bf v; uint4 q[2]; } u;
    u.q[0] = *(const uint4*)(p);
    u.q[1] = *(const uint4*)(p + gap);
    return u.v;
}

// ---------------------------------------------------------------------------
// Kernel 1: codebook f32 -> bf16 hi/lo split + row norms ||c_k||^2
// ---------------------------------------------------------------------------
__global__ __launch_bounds__(256) void vq_prep_codebook(
    const float* __restrict__ cb,
    unsigned short* __restrict__ cb_hi,
    unsigned short* __restrict__ cb_lo,
    float* __restrict__ cnorm)
{
    __shared__ float red[256];
    const int k = blockIdx.x;
    const int t = threadIdx.x;
    const float v = cb[k * CDIM + t];
    const unsigned short h = f32_to_bf16_rn(v);
    const unsigned short l = f32_to_bf16_rn(v - bf16_to_f32(h));
    cb_hi[k * CDIM + t] = h;
    cb_lo[k * CDIM + t] = l;
    red[t] = v * v;
    __syncthreads();
    for (int s = 128; s > 0; s >>= 1) {
        if (t < s) red[t] += red[t + s];
        __syncthreads();
    }
    if (t == 0) cnorm[k] = red[0];
}

// ---------------------------------------------------------------------------
// Kernel 2: distance matmul via split-bf16 WMMA + argmin over K
// One workgroup = 16 query rows (same b,h; 16 consecutive w). 8 waves,
// each wave owns 8 of the 64 K-tiles (16 codes each).
// ---------------------------------------------------------------------------
__global__ __launch_bounds__(256) void vq_argmin_kernel(
    const float* __restrict__ z_e,             // [32,256,32,32] f32
    const unsigned short* __restrict__ cb_hi,  // [1024,256] bf16 bits
    const unsigned short* __restrict__ cb_lo,  // [1024,256] bf16 bits
    const float* __restrict__ cnorm,           // [1024]
    int* __restrict__ idx_out,                 // [32768]
    float* __restrict__ idx_f_out)             // [32768] (float copy for d_out)
{
    constexpr int AS = 264;                    // padded row stride (bank-conflict-free)
    __shared__ unsigned short s_zhi[16 * AS];
    __shared__ unsigned short s_zlo[16 * AS];
    __shared__ float s_cnorm[KCODES];
    __shared__ float s_rd[8 * 16];
    __shared__ int   s_rk[8 * 16];

    const int tid = threadIdx.x;
    const int n0  = blockIdx.x * 16;           // first query row of this WG
    const int bI  = n0 >> 10;                  // / (H*W)
    const int hI  = (n0 >> 5) & 31;
    const int w0  = n0 & 31;                   // 0 or 16

    // ---- stage 16 z rows -> LDS as bf16 hi/lo (coalesced 64B segments) ----
    {
        const int tw  = tid & 15;              // w within tile == LDS row
        const int tcg = tid >> 4;              // c group
        #pragma unroll
        for (int j = 0; j < 16; ++j) {
            const int c = tcg * 16 + j;
            const float v = z_e[(((bI * 256 + c) * 32 + hI) << 5) + w0 + tw];
            const unsigned short h = f32_to_bf16_rn(v);
            s_zhi[tw * AS + c] = h;
            s_zlo[tw * AS + c] = f32_to_bf16_rn(v - bf16_to_f32(h));
        }
    }
    for (int i = tid; i < KCODES; i += 256) s_cnorm[i] = cnorm[i];
    __syncthreads();

    const int wave   = tid >> 5;
    const int lane   = tid & 31;
    const int col    = lane & 15;              // k-column within tile / A row (M)
    const bool hiH   = lane >= 16;
    const int aoff   = hiH ? 8 : 0;            // A: lanes>=16 hold K {8..15,24..31}
    const int boff   = hiH ? 8 : 0;            // B: lanes>=16 hold K {8..15,24..31}

    const unsigned short* zh_row = s_zhi + col * AS;
    const unsigned short* zl_row = s_zlo + col * AS;

    float bestd[8];
    int   bestk[8];
    #pragma unroll
    for (int r = 0; r < 8; ++r) { bestd[r] = 3.4e38f; bestk[r] = 0x7fffffff; }

    for (int t = wave; t < 64; t += 8) {       // K-tile loop (wave-uniform)
        v8f acc = {};
        const int kg = t * 16 + col;           // global code index for this lane
        const unsigned short* bh = cb_hi + kg * CDIM + boff;
        const unsigned short* bl = cb_lo + kg * CDIM + boff;
        #pragma unroll
        for (int ch = 0; ch < 8; ++ch) {       // 8 chunks of K=32 over C=256
            const int kb = ch * 32;
            v16bf ah  = load_frag(zh_row + kb + aoff, 16);
            v16bf al  = load_frag(zl_row + kb + aoff, 16);
            v16bf bhf = load_frag(bh + kb, 16);
            v16bf blf = load_frag(bl + kb, 16);
            // dot ≈ zh*ch + zh*cl + zl*ch  (same f32 accumulator)
            acc = __builtin_amdgcn_wmma_f32_16x16x32_bf16(false, ah, false, bhf,
                                                          (short)0, acc, false, false);
            acc = __builtin_amdgcn_wmma_f32_16x16x32_bf16(false, ah, false, blf,
                                                          (short)0, acc, false, false);
            acc = __builtin_amdgcn_wmma_f32_16x16x32_bf16(false, al, false, bhf,
                                                          (short)0, acc, false, false);
        }
        const float cn = s_cnorm[kg];
        #pragma unroll
        for (int r = 0; r < 8; ++r) {
            const float d = fmaf(-2.0f, acc[r], cn);   // ||z||^2 const per row: dropped
            if (d < bestd[r] || (d == bestd[r] && kg < bestk[r])) {
                bestd[r] = d; bestk[r] = kg;
            }
        }
    }

    // ---- argmin across the 16 lanes holding different k for the same rows ----
    #pragma unroll
    for (int m = 8; m >= 1; m >>= 1) {
        #pragma unroll
        for (int r = 0; r < 8; ++r) {
            const float od = __shfl_xor(bestd[r], m, 32);
            const int   ok = __shfl_xor(bestk[r], m, 32);
            if (od < bestd[r] || (od == bestd[r] && ok < bestk[r])) {
                bestd[r] = od; bestk[r] = ok;
            }
        }
    }
    if (col == 0) {                             // lane 0 -> rows 0..7, lane 16 -> rows 8..15
        const int rbase = hiH ? 8 : 0;
        #pragma unroll
        for (int r = 0; r < 8; ++r) {
            s_rd[wave * 16 + rbase + r] = bestd[r];
            s_rk[wave * 16 + rbase + r] = bestk[r];
        }
    }
    __syncthreads();
    if (tid < 16) {                             // merge across 8 waves
        float bd = s_rd[tid];
        int   bk = s_rk[tid];
        #pragma unroll
        for (int wv = 1; wv < 8; ++wv) {
            const float od = s_rd[wv * 16 + tid];
            const int   ok = s_rk[wv * 16 + tid];
            if (od < bd || (od == bd && ok < bk)) { bd = od; bk = ok; }
        }
        idx_out[n0 + tid]   = bk;
        idx_f_out[n0 + tid] = (float)bk;
    }
}

// ---------------------------------------------------------------------------
// Kernel 3: gather codebook rows into the reference's permuted [B,W,C,H] output
// out[((bb*32+ww)*256+cc)*32+hh] = codebook[idx[src>>8]][src&255],
// src = ((bb*256+cc)*32+hh)*32+ww   (raw reinterpret + permute, per reference)
// ---------------------------------------------------------------------------
__global__ __launch_bounds__(256) void vq_gather_out(
    const float* __restrict__ cb,
    const int* __restrict__ idx,
    float* __restrict__ out)
{
    const int o  = blockIdx.x * 256 + threadIdx.x;
    const int hh = o & 31;
    const int cc = (o >> 5) & 255;
    const int ww = (o >> 13) & 31;
    const int bb = o >> 18;
    const int src = (((bb * 256 + cc) * 32 + hh) << 5) + ww;
    out[o] = cb[idx[src >> 8] * CDIM + (src & 255)];
}

extern "C" void kernel_launch(void* const* d_in, const int* in_sizes, int n_in,
                              void* d_out, int out_size, void* d_ws, size_t ws_size,
                              hipStream_t stream) {
    const float* z_e = (const float*)d_in[0];   // [32,256,32,32]
    const float* cb  = (const float*)d_in[1];   // [1024,256]

    // workspace partition (≈1.13 MB total)
    unsigned short* cb_hi = (unsigned short*)d_ws;                       // 512 KB
    unsigned short* cb_lo = cb_hi + (size_t)KCODES * CDIM;               // 512 KB
    float* cnorm = (float*)((char*)d_ws + 1024 * 1024);                  // 4 KB
    int*   idx   = (int*)((char*)d_ws + 1024 * 1024 + 4096);             // 128 KB

    float* out_zq  = (float*)d_out;             // 8388608 f32
    float* out_idx = out_zq + ZQ_ELEMS;         // 32768 indices as f32

    vq_prep_codebook<<<KCODES, 256, 0, stream>>>(cb, cb_hi, cb_lo, cnorm);
    vq_argmin_kernel<<<NVEC / 16, 256, 0, stream>>>(z_e, cb_hi, cb_lo, cnorm,
                                                    idx, out_idx);
    vq_gather_out<<<ZQ_ELEMS / 256, 256, 0, stream>>>(cb, idx, out_zq);
}
